// DraftAttention_82643760709672
// MI455X (gfx1250) — matrix operand order repaired and checked by reading, not compiled, not run on hardware
//
#include <hip/hip_runtime.h>
#include <hip/hip_bf16.h>

// ---------------------------------------------------------------------------
// Problem constants (B=2, S=2048, H=4096, NH=32, NKV=8, HD=128)
// ---------------------------------------------------------------------------
namespace {
constexpr int kB   = 2;
constexpr int kS   = 2048;
constexpr int kH   = 4096;
constexpr int kNH  = 32;
constexpr int kNKV = 8;
constexpr int kHD  = 128;
constexpr int kM   = kB * kS;  // 4096 activation rows
}

typedef __bf16        v16bf __attribute__((ext_vector_type(16)));
typedef float         v8f   __attribute__((ext_vector_type(8)));
typedef unsigned int  v4u   __attribute__((ext_vector_type(4)));

union Frag {
  v4u   q[2];   // two 16-byte chunks
  v16bf v;      // one WMMA bf16 A/B fragment
};

__device__ __forceinline__ v8f wmma_bf16(v16bf a, v16bf b, v8f c) {
  // D = A(16x32) * B(32x16) + C ; emits v_wmma_f32_16x16x32_bf16
  return __builtin_amdgcn_wmma_f32_16x16x32_bf16(
      /*neg_a=*/false, a, /*neg_b=*/false, b,
      /*c_mod=*/(short)0, c, /*reuse_a=*/false, /*reuse_b=*/false);
}

__device__ __forceinline__ v8f v8f_zero() {
  v8f r;
#pragma unroll
  for (int i = 0; i < 8; ++i) r[i] = 0.0f;
  return r;
}

__device__ __forceinline__ void cstore(float* p, float v)  { *p = v; }
__device__ __forceinline__ void cstore(__bf16* p, float v) { *p = (__bf16)v; }

// Async copy of one 16B chunk: global -> LDS, tracked by ASYNCcnt (no VGPR data).
__device__ __forceinline__ void async_b128(const __bf16* gsrc, __bf16* lds_dst) {
  // LDS aperture: generic shared address low 32 bits == LDS byte address.
  unsigned int lds = (unsigned int)(unsigned long long)(uintptr_t)lds_dst;
  unsigned long long ga = (unsigned long long)(uintptr_t)gsrc;
  asm volatile("global_load_async_to_lds_b128 %0, %1, off"
               :: "v"(lds), "v"(ga) : "memory");
}
__device__ __forceinline__ void wait_async0() {
  asm volatile("s_wait_asynccnt 0x0" ::: "memory");
}

// ---------------------------------------------------------------------------
// Elementwise helpers
// ---------------------------------------------------------------------------
__global__ void cvt_f32_to_bf16(const float* __restrict__ in,
                                __bf16* __restrict__ out, long long n) {
  long long i = (long long)blockIdx.x * blockDim.x + threadIdx.x;
  if (i < n) out[i] = (__bf16)in[i];
}

// W (K x N, fp32, row-major) -> WT (N x K, bf16, row-major)
__global__ void transpose_cvt(const float* __restrict__ W,
                              __bf16* __restrict__ WT, int K, int N) {
  long long i = (long long)blockIdx.x * blockDim.x + threadIdx.x;
  long long total = (long long)K * N;
  if (i >= total) return;
  int n = (int)(i / K);
  int k = (int)(i % K);
  WT[i] = (__bf16)W[(size_t)k * N + n];
}

// X: (B*S, nh*HD) bf16 -> Y: [B][nh][S][HD] bf16, RoPE applied
__global__ void rope_kernel(const __bf16* __restrict__ X,
                            __bf16* __restrict__ Y, int nh, long long total) {
  long long i = (long long)blockIdx.x * blockDim.x + threadIdx.x;
  if (i >= total) return;
  int d = (int)(i & (kHD - 1));
  long long t = i >> 7;                 // HD = 128 = 2^7
  int pos  = (int)(t % kS);  t /= kS;
  int head = (int)(t % nh);
  int b    = (int)(t / nh);

  int j = d & 63;                       // freq index (d % (HD/2))
  float inv = __expf(-((float)(2 * j) / 128.0f) * 9.210340371976184f); // ln(1e4)
  float ang = (float)pos * inv;
  float c = __cosf(ang), s = __sinf(ang);

  size_t base = (size_t)(b * kS + pos) * (size_t)(nh * kHD) + (size_t)head * kHD;
  float x  = (float)X[base + d];
  float xr = (float)X[base + (d < 64 ? d + 64 : d - 64)];
  float out = (d < 64) ? (x * c - xr * s) : (x * c + xr * s);
  Y[i] = (__bf16)out;
}

// V: (B*S, NKV*HD) bf16 -> VT: [B][NKV][HD][S] bf16
__global__ void v_transpose(const __bf16* __restrict__ V,
                            __bf16* __restrict__ VT, long long total) {
  long long i = (long long)blockIdx.x * blockDim.x + threadIdx.x;
  if (i >= total) return;
  int pos = (int)(i % kS);  long long t = i / kS;
  int d   = (int)(t % kHD); t /= kHD;
  int kv  = (int)(t % kNKV);
  int b   = (int)(t / kNKV);
  VT[i] = V[(size_t)(b * kS + pos) * (size_t)(kNKV * kHD) + (size_t)kv * kHD + d];
}

// ---------------------------------------------------------------------------
// bf16 WMMA GEMM with async-LDS double buffering:
//   C(MxN) = A(MxK, row-major) * BT(NxK, row-major)^T
// Block: 256 threads = 8 waves, tile 128(M) x 128(N); wave tile 32x64.
// Per K-step: 128x32 A and B tiles staged in LDS via
// global_load_async_to_lds_b128 (ASYNCcnt), then waves read WMMA fragments
// with ds_load_b128 in the ISA lane layout:
//   A lane: 16B chunks at k = 8h and 16+8h of its row
//   B lane: 16B chunks at k = 8h and 16+8h of its column (row of BT)
// ---------------------------------------------------------------------------
template <typename OutT>
__global__ __launch_bounds__(256) void gemm_bf16_nt(
    const __bf16* __restrict__ A, const __bf16* __restrict__ BT,
    OutT* __restrict__ C, int M, int N, int K) {
  __shared__ __bf16 Ash[2][128 * 32];   // [buf][row*32 + k]  (8 KB each)
  __shared__ __bf16 Bsh[2][128 * 32];

  const int tid  = threadIdx.x;
  const int lane = tid & 31;
  const int wave = tid >> 5;
  const int h    = lane >> 4;
  const int l16  = lane & 15;
  const int wm   = wave >> 1;                 // 0..3
  const int wn   = wave & 1;                  // 0..1
  const int mblk = blockIdx.x * 128;          // block row base
  const int nblk = blockIdx.y * 128;          // block col base (rows of BT)
  const int m0   = mblk + wm * 32;
  const int n0   = nblk + wn * 64;
  (void)M;

  // tile loader: 128 rows x 32 cols; thread t covers quarter (t&3) of rows
  // t>>2 and t>>2 + 64  -> 2 async b128 per thread per tile
  const int lrow = tid >> 2;
  const int lq   = tid & 3;
  auto issue_tile = [&](const __bf16* src, int row0, int k, __bf16* dst) {
#pragma unroll
    for (int rr = 0; rr < 2; ++rr) {
      const int row = lrow + rr * 64;
      async_b128(src + (size_t)(row0 + row) * K + k + lq * 8,
                 dst + row * 32 + lq * 8);
    }
  };

  v8f acc[2][4];
#pragma unroll
  for (int mf = 0; mf < 2; ++mf)
#pragma unroll
    for (int nf = 0; nf < 4; ++nf) acc[mf][nf] = v8f_zero();

  int cur = 0;
  issue_tile(A,  mblk, 0, Ash[0]);
  issue_tile(BT, nblk, 0, Bsh[0]);

  for (int k = 0; k < K; k += 32) {
    wait_async0();        // own async loads for buffer `cur` landed in LDS
    __syncthreads();      // all waves' loads landed; prev compute finished

    if (k + 32 < K) {     // prefetch next K-slice into the other buffer
      issue_tile(A,  mblk, k + 32, Ash[cur ^ 1]);
      issue_tile(BT, nblk, k + 32, Bsh[cur ^ 1]);
    }

    Frag a[2];
#pragma unroll
    for (int mf = 0; mf < 2; ++mf) {
      const __bf16* ap = &Ash[cur][(wm * 32 + mf * 16 + l16) * 32];
      a[mf].q[0] = *(const v4u*)(ap + 8 * h);
      a[mf].q[1] = *(const v4u*)(ap + 16 + 8 * h);
    }
    Frag b[4];
#pragma unroll
    for (int nf = 0; nf < 4; ++nf) {
      const __bf16* bp = &Bsh[cur][(wn * 64 + nf * 16 + l16) * 32 + 8 * h];
      b[nf].q[0] = *(const v4u*)(bp);
      b[nf].q[1] = *(const v4u*)(bp + 16);
    }
#pragma unroll
    for (int mf = 0; mf < 2; ++mf)
#pragma unroll
      for (int nf = 0; nf < 4; ++nf)
        acc[mf][nf] = wmma_bf16(a[mf].v, b[nf].v, acc[mf][nf]);

    cur ^= 1;
  }

#pragma unroll
  for (int mf = 0; mf < 2; ++mf)
#pragma unroll
    for (int nf = 0; nf < 4; ++nf)
#pragma unroll
      for (int v = 0; v < 8; ++v) {
        int r = m0 + mf * 16 + v + 8 * h;
        int c = n0 + nf * 16 + l16;
        cstore(&C[(size_t)r * N + c], acc[mf][nf][v]);
      }
}

// ---------------------------------------------------------------------------
// Flash attention (causal, GQA 4:1), bf16 WMMA, online softmax.
// Grid: (B*NH, S/128). Block: 256 threads = 8 waves; each wave owns a
// 16-query-row tile and loops over 32-key blocks up to the causal frontier.
// All 8 K-fragments (and V-fragments) are preloaded into distinct registers
// before their WMMA bursts so load->WMMA chains can pipeline.
// ---------------------------------------------------------------------------
__global__ __launch_bounds__(256) void flash_attn(
    const __bf16* __restrict__ Q,   // [B][NH][S][HD]
    const __bf16* __restrict__ Kr,  // [B][NKV][S][HD]
    const __bf16* __restrict__ VT,  // [B][NKV][HD][S]
    __bf16* __restrict__ O)         // [B*S][NH*HD]
{
  __shared__ __bf16 plds[8][16 * 32];   // wave-private P staging (C->A relayout)

  const int lane = threadIdx.x & 31;
  const int wave = threadIdx.x >> 5;
  const int h    = lane >> 4;
  const int l16  = lane & 15;
  const int bh   = blockIdx.x;
  const int b    = bh / kNH;
  const int head = bh % kNH;
  const int kv   = head / (kNH / kNKV);
  const int qbase = (blockIdx.y * 8 + wave) * 16;

  const __bf16* Qb = Q  + (size_t)bh * kS * kHD;
  const __bf16* Kb = Kr + (size_t)(b * kNKV + kv) * kS * kHD;
  const __bf16* Vb = VT + (size_t)(b * kNKV + kv) * kHD * kS;

  // Q fragments for this 16-row tile (K-dim = HD = 128 -> 4 chunks of 32)
  Frag aq[4];
  {
    const __bf16* qp = Qb + (size_t)(qbase + l16) * kHD;
#pragma unroll
    for (int j = 0; j < 4; ++j) {
      aq[j].q[0] = *(const v4u*)(qp + 32 * j + 8 * h);
      aq[j].q[1] = *(const v4u*)(qp + 32 * j + 16 + 8 * h);
    }
  }

  v8f oacc[8];
#pragma unroll
  for (int nb = 0; nb < 8; ++nb) oacc[nb] = v8f_zero();
  v8f mrow, lrow;
#pragma unroll
  for (int v = 0; v < 8; ++v) { mrow[v] = -3.0e38f; lrow[v] = 0.0f; }

  const float scale = 0.08838834764831845f;  // 1/sqrt(128)
  const int kb_max = (qbase + 15) >> 5;      // last 32-key block (causal)

  for (int kb = 0; kb <= kb_max; ++kb) {
    const int kc0 = kb * 32;

    // speculative prefetch of next K/V blocks (global_prefetch_b8)
    if (kb < kb_max) {
      __builtin_prefetch(Kb + (size_t)(kc0 + 32 + l16) * kHD + 16 * h, 0, 1);
      __builtin_prefetch(Vb + (size_t)(wave * 16 + l16) * kS + kc0 + 32 + 16 * h, 0, 1);
    }

    // ---- preload all 8 K-fragments (independent live ranges)
    Frag bk[8];
#pragma unroll
    for (int j = 0; j < 4; ++j) {
      const __bf16* kp0 = Kb + (size_t)(kc0 + l16) * kHD + 32 * j + 8 * h;
      const __bf16* kp1 = Kb + (size_t)(kc0 + 16 + l16) * kHD + 32 * j + 8 * h;
      bk[j].q[0]     = *(const v4u*)kp0;  bk[j].q[1]     = *(const v4u*)(kp0 + 16);
      bk[4 + j].q[0] = *(const v4u*)kp1;  bk[4 + j].q[1] = *(const v4u*)(kp1 + 16);
    }

    // ---- scores: two 16-col fragments, accumulated over HD in 4 WMMAs each
    v8f s0 = v8f_zero(), s1 = v8f_zero();
#pragma unroll
    for (int j = 0; j < 4; ++j) {
      s0 = wmma_bf16(aq[j].v, bk[j].v, s0);
      s1 = wmma_bf16(aq[j].v, bk[4 + j].v, s1);
    }

    // ---- scale + causal mask + online softmax (rows live in 16-lane halves)
#pragma unroll
    for (int v = 0; v < 8; ++v) {
      const int row = qbase + v + 8 * h;
      float x0 = s0[v] * scale;  if (kc0 + l16 > row)      x0 = -3.0e38f;
      float x1 = s1[v] * scale;  if (kc0 + 16 + l16 > row) x1 = -3.0e38f;

      float mx = fmaxf(x0, x1);
      mx = fmaxf(mx, __shfl_xor(mx, 1));
      mx = fmaxf(mx, __shfl_xor(mx, 2));
      mx = fmaxf(mx, __shfl_xor(mx, 4));
      mx = fmaxf(mx, __shfl_xor(mx, 8));

      const float mold = mrow[v];
      const float mnew = fmaxf(mold, mx);
      const float alpha = __expf(mold - mnew);
      const float p0 = __expf(x0 - mnew);
      const float p1 = __expf(x1 - mnew);

      float rs = p0 + p1;
      rs += __shfl_xor(rs, 1);
      rs += __shfl_xor(rs, 2);
      rs += __shfl_xor(rs, 4);
      rs += __shfl_xor(rs, 8);

      lrow[v] = lrow[v] * alpha + rs;
      mrow[v] = mnew;
#pragma unroll
      for (int nb = 0; nb < 8; ++nb) oacc[nb][v] *= alpha;

      // P (C-layout) -> LDS, bf16
      plds[wave][(v + 8 * h) * 32 + l16]      = (__bf16)p0;
      plds[wave][(v + 8 * h) * 32 + 16 + l16] = (__bf16)p1;
    }
    asm volatile("s_wait_dscnt 0" ::: "memory");

    // ---- reload P in A-fragment layout
    Frag pf;
    {
      const __bf16* pp = &plds[wave][l16 * 32 + 8 * h];
      pf.q[0] = *(const v4u*)(pp);
      pf.q[1] = *(const v4u*)(pp + 16);
    }

    // ---- preload all 8 V-fragments, then O += P * V
    Frag vf[8];
#pragma unroll
    for (int nb = 0; nb < 8; ++nb) {
      const __bf16* vp = Vb + (size_t)(nb * 16 + l16) * kS + kc0 + 8 * h;
      vf[nb].q[0] = *(const v4u*)vp;  vf[nb].q[1] = *(const v4u*)(vp + 16);
    }
#pragma unroll
    for (int nb = 0; nb < 8; ++nb)
      oacc[nb] = wmma_bf16(pf.v, vf[nb].v, oacc[nb]);
  }

  // ---- epilogue: normalize and store row-major [B*S][NH*HD]
#pragma unroll
  for (int nb = 0; nb < 8; ++nb)
#pragma unroll
    for (int v = 0; v < 8; ++v) {
      const int row = qbase + v + 8 * h;
      const float val = oacc[nb][v] / lrow[v];
      O[(size_t)(b * kS + row) * (kNH * kHD) + (size_t)head * kHD + nb * 16 + l16] =
          (__bf16)val;
    }
}

// ---------------------------------------------------------------------------
// Host orchestration
// ---------------------------------------------------------------------------
extern "C" void kernel_launch(void* const* d_in, const int* in_sizes, int n_in,
                              void* d_out, int out_size, void* d_ws, size_t ws_size,
                              hipStream_t stream) {
  (void)in_sizes; (void)n_in; (void)out_size; (void)ws_size;
  const float* hidden = (const float*)d_in[0];
  // d_in[1] = position_ids (deterministic 0..S-1; recomputed on device)
  const float* Wq = (const float*)d_in[2];
  const float* Wk = (const float*)d_in[3];
  const float* Wv = (const float*)d_in[4];
  const float* Wo = (const float*)d_in[5];

  char* ws = (char*)d_ws;
  size_t off = 0;
  auto take = [&](size_t bytes) -> char* {
    char* p = ws + off;
    off += (bytes + 255) & ~(size_t)255;
    return p;
  };

  __bf16* hbf = (__bf16*)take((size_t)kM * kH * 2);
  __bf16* WqT = (__bf16*)take((size_t)kH * kH * 2);
  __bf16* WkT = (__bf16*)take((size_t)(kNKV * kHD) * kH * 2);
  __bf16* WvT = (__bf16*)take((size_t)(kNKV * kHD) * kH * 2);
  __bf16* WoT = (__bf16*)take((size_t)kH * kH * 2);
  __bf16* Qp  = (__bf16*)take((size_t)kM * (kNH * kHD) * 2);
  __bf16* Kp  = (__bf16*)take((size_t)kM * (kNKV * kHD) * 2);
  __bf16* Vp  = (__bf16*)take((size_t)kM * (kNKV * kHD) * 2);
  __bf16* Qr  = (__bf16*)take((size_t)kB * kNH * kS * kHD * 2);
  __bf16* Kr  = (__bf16*)take((size_t)kB * kNKV * kS * kHD * 2);
  __bf16* Vt  = (__bf16*)take((size_t)kB * kNKV * kHD * kS * 2);
  __bf16* Ob  = (__bf16*)take((size_t)kM * (kNH * kHD) * 2);

  // 1) hidden fp32 -> bf16
  {
    long long n = (long long)kM * kH;
    cvt_f32_to_bf16<<<(unsigned)((n + 255) / 256), 256, 0, stream>>>(hidden, hbf, n);
  }
  // 2) weight transposes (one-time, amortized)
  {
    long long nq = (long long)kH * kH;
    transpose_cvt<<<(unsigned)((nq + 255) / 256), 256, 0, stream>>>(Wq, WqT, kH, kH);
    transpose_cvt<<<(unsigned)((nq + 255) / 256), 256, 0, stream>>>(Wo, WoT, kH, kH);
    long long nk = (long long)kH * (kNKV * kHD);
    transpose_cvt<<<(unsigned)((nk + 255) / 256), 256, 0, stream>>>(Wk, WkT, kH, kNKV * kHD);
    transpose_cvt<<<(unsigned)((nk + 255) / 256), 256, 0, stream>>>(Wv, WvT, kH, kNKV * kHD);
  }
  // 3) QKV projections (WMMA bf16, async-LDS double-buffered)
  gemm_bf16_nt<__bf16><<<dim3(kM / 128, (kNH * kHD) / 128), 256, 0, stream>>>(
      hbf, WqT, Qp, kM, kNH * kHD, kH);
  gemm_bf16_nt<__bf16><<<dim3(kM / 128, (kNKV * kHD) / 128), 256, 0, stream>>>(
      hbf, WkT, Kp, kM, kNKV * kHD, kH);
  gemm_bf16_nt<__bf16><<<dim3(kM / 128, (kNKV * kHD) / 128), 256, 0, stream>>>(
      hbf, WvT, Vp, kM, kNKV * kHD, kH);
  // 4) RoPE + reshape to [B][h][S][HD]
  {
    long long nq = (long long)kB * kNH * kS * kHD;
    rope_kernel<<<(unsigned)((nq + 255) / 256), 256, 0, stream>>>(Qp, Qr, kNH, nq);
    long long nk = (long long)kB * kNKV * kS * kHD;
    rope_kernel<<<(unsigned)((nk + 255) / 256), 256, 0, stream>>>(Kp, Kr, kNKV, nk);
    // 5) V -> [B][NKV][HD][S]
    v_transpose<<<(unsigned)((nk + 255) / 256), 256, 0, stream>>>(Vp, Vt, nk);
  }
  // 6) causal flash attention
  flash_attn<<<dim3(kB * kNH, kS / 128), 256, 0, stream>>>(Qr, Kr, Vt, Ob);
  // 7) output projection -> fp32 d_out
  gemm_bf16_nt<float><<<dim3(kM / 128, kH / 128), 256, 0, stream>>>(
      Ob, WoT, (float*)d_out, kM, kH, kH);
}
